// NILReg_8744553415076
// MI455X (gfx1250) — matrix operand order repaired and checked by reading, not compiled, not run on hardware
//
#include <hip/hip_runtime.h>

// CDNA5 / gfx1250, wave32.
typedef __attribute__((ext_vector_type(8)))  float    v8f;
typedef __attribute__((ext_vector_type(8)))  unsigned v8u;
typedef __attribute__((ext_vector_type(16))) __bf16   v16bf;

#define M_NODES 4096             // num_nodes (power of two; reference does % m)
#define T_DIM   32               // time dim of y_hat
#define WPR     (M_NODES / 32)   // bitmap words per adjacency row = 128

__device__ __forceinline__ unsigned short f32_to_bf16_rne(float f) {
  unsigned u = __builtin_bit_cast(unsigned, f);
  unsigned r = u + 0x7FFFu + ((u >> 16) & 1u);
  return (unsigned short)(r >> 16);
}
__device__ __forceinline__ float bf16_to_f32(unsigned short h) {
  unsigned u = ((unsigned)h) << 16;
  return __builtin_bit_cast(float, u);
}

// ---------------------------------------------------------------------------
// Kernel 1: zero the adjacency bitmap (16 MB, L2-resident) and the scalar out.
// ---------------------------------------------------------------------------
__global__ void nilreg_zero(unsigned* __restrict__ bitmap, size_t nwords,
                            float* __restrict__ out) {
  size_t i = (size_t)blockIdx.x * blockDim.x + threadIdx.x;
  size_t stride = (size_t)gridDim.x * blockDim.x;
  for (; i < nwords; i += stride) bitmap[i] = 0u;
  if (blockIdx.x == 0 && threadIdx.x == 0) out[0] = 0.0f;
}

// ---------------------------------------------------------------------------
// Kernel 2: transpose y_hat (b,k,t) -> K-contiguous bf16 split planes
// hi[b][t][k], lo[b][t][k] with yb = hi + lo to ~2^-17 relative accuracy.
// ---------------------------------------------------------------------------
__global__ void nilreg_transpose(const float* __restrict__ yhat,
                                 unsigned short* __restrict__ hi,
                                 unsigned short* __restrict__ lo,
                                 int total) {
  int tid = blockIdx.x * blockDim.x + threadIdx.x;
  if (tid >= total) return;
  int t  = tid & (T_DIM - 1);
  int bk = tid >> 5;                 // b*m + k   (T = 32)
  int b  = bk >> 12;                 // / m
  int k  = bk & (M_NODES - 1);
  float v = yhat[tid];
  unsigned short h = f32_to_bf16_rne(v);
  unsigned short l = f32_to_bf16_rne(v - bf16_to_f32(h));
  size_t o = ((size_t)b * T_DIM + t) * M_NODES + k;
  hi[o] = h;
  lo[o] = l;
}

// ---------------------------------------------------------------------------
// Kernel 3: scatter COO edges into the per-batch m x m bit matrix.
// atomicOr exactly reproduces jnp .set(1.0) dedup semantics.
// ---------------------------------------------------------------------------
__global__ void nilreg_scatter(const long long* __restrict__ coo,
                               unsigned* __restrict__ bitmap,
                               int E, int bs) {
  int tid = blockIdx.x * blockDim.x + threadIdx.x;
  int total = bs * E;
  if (tid >= total) return;
  int b = tid / E;
  int e = tid - b * E;
  const long long* base = coo + (size_t)b * 2 * E;
  int src = (int)base[e]     & (M_NODES - 1);
  int tgt = (int)base[E + e] & (M_NODES - 1);
  atomicOr(&bitmap[((size_t)b * M_NODES + src) * WPR + (tgt >> 5)],
           1u << (tgt & 31));
}

// ---------------------------------------------------------------------------
// Kernel 4: agg = (A_edges + I) @ yb[:, :31] via v_wmma_f32_16x16x32_bf16
// (split hi/lo B for fp32-grade accuracy), fused with
// reg = sum(relu(yb[:,1:] - agg)).  One wave per 16x16 output tile.
// ---------------------------------------------------------------------------
__global__ __launch_bounds__(256) void nilreg_spmm_relu_sum(
    const float* __restrict__ yhat,
    const unsigned* __restrict__ bitmap,
    const unsigned short* __restrict__ ybt_hi,
    const unsigned short* __restrict__ ybt_lo,
    float* __restrict__ out) {
  const int lane = threadIdx.x & 31;
  const int wave = threadIdx.x >> 5;

  // tile id: [b:3][rowtile:8][coltile:1] -> 4096 tiles total
  const int tile = blockIdx.x * 8 + wave;
  const int t0 = (tile & 1) * 16;            // output cols t0 .. t0+15
  const int i0 = ((tile >> 1) & 255) * 16;   // output rows i0 .. i0+15
  const int b  = tile >> 9;

  const float* yb = yhat + (size_t)b * M_NODES * T_DIM;
  // Each lane tracks the bitmap row for M = i0 + (lane & 15); b128 word reads.
  const uint4* rowBits4 = (const uint4*)(
      bitmap + ((size_t)b * M_NODES + (i0 + (lane & 15))) * WPR);

  const int col = t0 + (lane & 15);          // B/C/D column = N
  // A uses contiguous K order (matches B): lanes 0-15 own K=0..15; 16-31 own K=16..31.
  const int kb0 = (lane >= 16) ? 16 : 0;
  // B uses the same contiguous K order: lanes 0-15 hold K=0..15, lanes 16-31 K=16..31.
  const int kbB = (lane >= 16) ? 16 : 0;

  const __bf16* bcol_hi =
      (const __bf16*)(ybt_hi + ((size_t)b * T_DIM + col) * M_NODES);
  const __bf16* bcol_lo =
      (const __bf16*)(ybt_lo + ((size_t)b * T_DIM + col) * M_NODES);

  v8f c = {0.f, 0.f, 0.f, 0.f, 0.f, 0.f, 0.f, 0.f};

  for (int w4 = 0; w4 < WPR / 4; ++w4) {
    const uint4 wd = rowBits4[w4];
    if (!__any((wd.x | wd.y | wd.z | wd.w) != 0u)) continue;  // coarse skip
    const unsigned wv[4] = {wd.x, wd.y, wd.z, wd.w};
#pragma unroll
    for (int q = 0; q < 4; ++q) {
      const unsigned word = wv[q];
      if (!__any(word != 0u)) continue;      // fine-grained skip (wave-uniform)
      // --- build A fragment: bits -> packed bf16 {0,1} pairs ---
      v8u au;
#pragma unroll
      for (int j = 0; j < 4; ++j) {
        unsigned p0 = (word >> (kb0 + 2 * j)) & 1u;
        unsigned p1 = (word >> (kb0 + 2 * j + 1)) & 1u;
        au[j] = p0 * 0x3F80u | p1 * 0x3F800000u;          // K = kb0+2j, +1
        unsigned q0 = (word >> (kb0 + 8 + 2 * j)) & 1u;
        unsigned q1 = (word >> (kb0 + 8 + 2 * j + 1)) & 1u;
        au[j + 4] = q0 * 0x3F80u | q1 * 0x3F800000u;      // K = kb0+8+2j, +1
      }
      const v16bf a = __builtin_bit_cast(v16bf, au);
      // --- B fragments: 32B contiguous vector loads (K-major layout) ---
      const int kofs = (w4 * 4 + q) * 32 + kbB;
      const v16bf bh = *(const v16bf*)(bcol_hi + kofs);
      const v16bf bl = *(const v16bf*)(bcol_lo + kofs);
      // (neg_a, A, neg_b, B, c_mod, C, reuse_a, reuse_b)
      c = __builtin_amdgcn_wmma_f32_16x16x32_bf16(false, a, false, bh,
                                                  (short)0, c, false, false);
      c = __builtin_amdgcn_wmma_f32_16x16x32_bf16(false, a, false, bl,
                                                  (short)0, c, false, false);
    }
  }

  // Epilogue: identity term analytically (pure fp32), relu-diff, reduce.
  float sum = 0.0f;
#pragma unroll
  for (int r = 0; r < 8; ++r) {
    // 16x16 f32 C/D layout: VGPR r, lanes 0-15 -> M=r, lanes 16-31 -> M=r+8
    const int row = i0 + r + ((lane >= 16) ? 8 : 0);
    const float diag = yb[(size_t)row * T_DIM + col];     // + I @ yb
    const float agg  = c[r] + diag;
    if (col < T_DIM - 1) {                                // valid t = 0..30
      const float diff = yb[(size_t)row * T_DIM + col + 1] - agg;
      sum += (diff > 0.0f) ? diff : 0.0f;
    }
  }
#pragma unroll
  for (int off = 16; off > 0; off >>= 1) sum += __shfl_down(sum, off, 32);
  if (lane == 0) atomicAdd(out, sum);
}

// ---------------------------------------------------------------------------
extern "C" void kernel_launch(void* const* d_in, const int* in_sizes, int n_in,
                              void* d_out, int out_size, void* d_ws,
                              size_t ws_size, hipStream_t stream) {
  const int m = M_NODES, T = T_DIM;
  const int bs = in_sizes[0] / (m * T);       // 8
  const int E  = in_sizes[1] / (bs * 2);      // 65536

  const float* yhat    = (const float*)d_in[0];
  const long long* coo = (const long long*)d_in[1];   // int64 COO

  float* out = (float*)d_out;

  // Workspace layout: [bitmap 16MB][ybt_hi 2MB][ybt_lo 2MB]
  unsigned* bitmap = (unsigned*)d_ws;
  const size_t nwords = (size_t)bs * m * WPR;
  unsigned short* ybt_hi = (unsigned short*)((char*)d_ws + nwords * 4);
  unsigned short* ybt_lo = ybt_hi + (size_t)bs * T * m;

  nilreg_zero<<<1024, 256, 0, stream>>>(bitmap, nwords, out);

  const int total_y = bs * m * T;
  nilreg_transpose<<<(total_y + 255) / 256, 256, 0, stream>>>(yhat, ybt_hi,
                                                              ybt_lo, total_y);

  const int total_edges = bs * E;
  nilreg_scatter<<<(total_edges + 255) / 256, 256, 0, stream>>>(coo, bitmap,
                                                                E, bs);

  const int tiles = bs * (m / 16) * 2;        // 4096 tiles, 8 waves/block
  nilreg_spmm_relu_sum<<<tiles / 8, 256, 0, stream>>>(yhat, bitmap, ybt_hi,
                                                      ybt_lo, out);
}
